// DecomposedQValueNN_5798205849722
// MI455X (gfx1250) — compile-verified, matrix operand order checked by reading, not run on hardware
//
#include <hip/hip_runtime.h>
#include <hip/hip_bf16.h>

typedef __attribute__((ext_vector_type(16))) _Float16 v16h;
typedef __attribute__((ext_vector_type(8)))  _Float16 v8h;
typedef __attribute__((ext_vector_type(8)))  float    v8f;

#define DEV static __device__ __forceinline__

// K index held by (lane, vgpr v) for 16-bit A (16xK) and B (Kx16) operands.
DEV int kbase(int lane, int v) {
  return ((v & 4) ? 16 : 0) + ((lane & 16) ? 8 : 0) + 2 * (v & 3);
}

DEV v8f wmma16(v16h a, v16h b, v8f c) {
  // (neg_a, A, neg_b, B, c_mod, C, reuse_a, reuse_b)
  return __builtin_amdgcn_wmma_f32_16x16x32_f16(false, a, false, b, (short)0, c,
                                                false, false);
}

// Wave-local LDS ordering: DS ops are in-order per wave; this is a compiler
// barrier + explicit CDNA5 split-counter wait.
DEV void lds_fence() { asm volatile("s_wait_dscnt 0x0" ::: "memory"); }

// Single-instruction relu. Plain fmaxf(x,0) lowers as canonicalize(max x,x)
// + max, and fmed3(x,0,inf) gets combined back to the same thing; inline asm
// bypasses the canonicalization so relu is exactly one v_max_num_f32.
DEV float relu1(float x) {
  float y;
  asm("v_max_num_f32 %0, 0, %1" : "=v"(y) : "v"(x));
  return y;
}

DEV v16h zero16() {
  v16h a;
#pragma unroll
  for (int i = 0; i < 16; ++i) a[i] = (_Float16)0.f;
  return a;
}

DEV v8f zero8() {
  v8f c;
#pragma unroll
  for (int i = 0; i < 8; ++i) c[i] = 0.f;
  return c;
}

// Generic 32x16 B-tile (K zero-padded to 32) from row-major f32 W[kmax][ldw].
DEV v16h load_Btile(const float* W, int ldw, int kmax, int nvalid, int nbase,
                    int lane) {
  v16h r;
  int n = nbase + (lane & 15);
  bool nok = n < nvalid;
#pragma unroll
  for (int v = 0; v < 8; ++v) {
    int k0 = kbase(lane, v);
    float w0 = (nok && (k0 < kmax)) ? W[k0 * ldw + n] : 0.f;
    float w1 = (nok && (k0 + 1 < kmax)) ? W[(k0 + 1) * ldw + n] : 0.f;
    r[2 * v]     = (_Float16)w0;
    r[2 * v + 1] = (_Float16)w1;
  }
  return r;
}

// Layer-1 B-tile: 4x32 weights in rows K=0..3, bias vector in row K=4
// (matched by a constant-1 at K=4 in the A operand) -> bias folded into GEMM.
DEV v16h load_Btile_l1(const float* W, const float* b, int nbase, int lane) {
  v16h r;
  int n = nbase + (lane & 15);
#pragma unroll
  for (int v = 0; v < 8; ++v) {
    int k0 = kbase(lane, v);
    float w0 = (k0 < 4) ? W[k0 * 32 + n] : (k0 == 4 ? b[n] : 0.f);
    float w1 = (k0 + 1 < 4) ? W[(k0 + 1) * 32 + n] : (k0 + 1 == 4 ? b[n] : 0.f);
    r[2 * v]     = (_Float16)w0;
    r[2 * v + 1] = (_Float16)w1;
  }
  return r;
}

DEV float bias_n(const float* b, int nvalid, int nbase, int lane) {
  int n = nbase + (lane & 15);
  return (n < nvalid) ? b[n] : 0.f;
}

DEV void relu8(v8f& c) {
#pragma unroll
  for (int i = 0; i < 8; ++i) c[i] = relu1(c[i]);
}

DEV void bias_relu(v8f& c, float b) {
#pragma unroll
  for (int i = 0; i < 8; ++i) c[i] = relu1(c[i] + b);
}

// C layout (16x16 f32): lane holds N = lane&15, rows M = v + 8*(lane>=16).
// Store as f16 into a row-major 16x32 staging tile at column base nbase.
DEV void store_C(_Float16* Hst, const v8f& c, int nbase, int lane) {
  int n = nbase + (lane & 15);
  int mb = (lane & 16) ? 8 : 0;
#pragma unroll
  for (int v = 0; v < 8; ++v) Hst[(mb + v) * 32 + n] = (_Float16)c[v];
}

// Read a 16x32 row-major f16 LDS tile back as a WMMA A operand
// (two ds_load_b128 per lane).
DEV v16h load_A(const _Float16* Hst, int lane) {
  int m = lane & 15;
  int kb = (lane & 16) ? 8 : 0;
  v8h lo = *(const v8h*)(Hst + m * 32 + kb);
  v8h hi = *(const v8h*)(Hst + m * 32 + 16 + kb);
  v16h a;
#pragma unroll
  for (int i = 0; i < 8; ++i) { a[i] = lo[i]; a[8 + i] = hi[i]; }
  return a;
}

// 4-wide input row tile (K padded 4->32), plus the constant-1 at K=4 that
// multiplies the bias row of the layer-1 B tile. Only lanes 0-15 nonzero.
DEV v16h load_A_in4(const float* base16rows, int lane) {
  v16h a = zero16();
  if (lane < 16) {
    float4 f = *(const float4*)(base16rows + lane * 4);
    a[0] = (_Float16)f.x; a[1] = (_Float16)f.y;
    a[2] = (_Float16)f.z; a[3] = (_Float16)f.w;
    a[4] = (_Float16)1.f;   // K=4 -> bias row
  }
  return a;
}

__global__ __launch_bounds__(256) void dqnn_kernel(
    const float* __restrict__ x, const int* __restrict__ selp,
    const float* oW1, const float* ob1, const float* oW2, const float* ob2,
    const float* oW3, const float* ob3, const float* sW1, const float* sb1,
    const float* sW2, const float* sb2, const float* sW3, const float* sb3,
    const float* gW1, const float* gb1, const float* gW2, const float* gb2,
    float* __restrict__ out, int Btot) {
  __shared__ __align__(16) _Float16 Hs[8][2][16 * 32];  // per-wave, per-chain
  __shared__ __align__(16) _Float16 Gs[8][16 * 32];     // concat(sel, sum_other)
  __shared__ int acts[8][16];

  const int lane = threadIdx.x & 31;
  const int w = threadIdx.x >> 5;
  const int s0 = (blockIdx.x * 8 + w) * 16;   // 16 samples per wave
  if (s0 + 16 > Btot) return;                 // wave-uniform exit

  const int sel = *selp;
  _Float16* HA = &Hs[w][0][0];
  _Float16* HB = &Hs[w][1][0];
  _Float16* Gsw = &Gs[w][0];
  const int mb = (lane & 16) ? 8 : 0;
  v8f z = zero8();

  // ---- selected-agent MLP first (its B tiles die before the main loop) ----
  {
    v16h W1B0 = load_Btile_l1(sW1, sb1, 0, lane);
    v16h W1B1 = load_Btile_l1(sW1, sb1, 16, lane);
    v16h W2B0 = load_Btile(sW2, 32, 32, 32, 0, lane);
    v16h W2B1 = load_Btile(sW2, 32, 32, 32, 16, lane);
    v16h W3B  = load_Btile(sW3, 16, 32, 16, 0, lane);
    float b2n0 = bias_n(sb2, 32, 0, lane), b2n1 = bias_n(sb2, 32, 16, lane);
    float b3n  = bias_n(sb3, 16, 0, lane);

    v16h as = zero16();
    if (lane < 16) {
      float4 f = *(const float4*)(x + (size_t)(s0 + lane) * 128 + sel * 4);
      as[0] = (_Float16)f.x; as[1] = (_Float16)f.y;
      as[2] = (_Float16)f.z; as[3] = (_Float16)f.w;
      as[4] = (_Float16)1.f;
      acts[w][lane] = (int)f.w;   // action index per sample
    }
    v8f c0 = wmma16(as, W1B0, z);
    v8f c1 = wmma16(as, W1B1, z);
    relu8(c0); relu8(c1);
    store_C(HA, c0, 0, lane); store_C(HA, c1, 16, lane);
    lds_fence();
    v16h a2 = load_A(HA, lane);
    c0 = wmma16(a2, W2B0, z);
    c1 = wmma16(a2, W2B1, z);
    bias_relu(c0, b2n0); bias_relu(c1, b2n1);
    lds_fence();
    store_C(HA, c0, 0, lane); store_C(HA, c1, 16, lane);
    lds_fence();
    v16h a3 = load_A(HA, lane);
    v8f so = wmma16(a3, W3B, z);
    bias_relu(so, b3n);
    lds_fence();
    store_C(Gsw, so, 0, lane);   // gating-input columns 0..15
  }

  // ---- other-agent MLP: both 16-agent halves of a sample as two
  //      independent WMMA chains sharing the B register tiles ----
  v16h W1B0 = load_Btile_l1(oW1, ob1, 0, lane);
  v16h W1B1 = load_Btile_l1(oW1, ob1, 16, lane);
  v16h W2B0 = load_Btile(oW2, 32, 32, 32, 0, lane);
  v16h W2B1 = load_Btile(oW2, 32, 32, 32, 16, lane);
  v16h W3B  = load_Btile(oW3, 16, 32, 16, 0, lane);
  float b2n0 = bias_n(ob2, 32, 0, lane), b2n1 = bias_n(ob2, 32, 16, lane);
  float b3n  = bias_n(ob3, 16, 0, lane);

#pragma unroll 1
  for (int sl = 0; sl < 16; ++sl) {
    const float* xs = x + (size_t)(s0 + sl) * 128;
    v16h a1a = load_A_in4(xs, lane);        // agents 0..15
    v16h a1b = load_A_in4(xs + 64, lane);   // agents 16..31

    // layer 1 (bias folded into GEMM)
    v8f c0a = wmma16(a1a, W1B0, z);
    v8f c0b = wmma16(a1b, W1B0, z);
    v8f c1a = wmma16(a1a, W1B1, z);
    v8f c1b = wmma16(a1b, W1B1, z);
    relu8(c0a); relu8(c0b); relu8(c1a); relu8(c1b);
    lds_fence();   // WAR vs previous iteration's ds_loads
    store_C(HA, c0a, 0, lane); store_C(HA, c1a, 16, lane);
    store_C(HB, c0b, 0, lane); store_C(HB, c1b, 16, lane);
    lds_fence();
    v16h a2a = load_A(HA, lane);
    v16h a2b = load_A(HB, lane);

    // layer 2
    c0a = wmma16(a2a, W2B0, z);
    c0b = wmma16(a2b, W2B0, z);
    c1a = wmma16(a2a, W2B1, z);
    c1b = wmma16(a2b, W2B1, z);
    bias_relu(c0a, b2n0); bias_relu(c0b, b2n0);
    bias_relu(c1a, b2n1); bias_relu(c1b, b2n1);
    lds_fence();
    store_C(HA, c0a, 0, lane); store_C(HA, c1a, 16, lane);
    store_C(HB, c0b, 0, lane); store_C(HB, c1b, 16, lane);
    lds_fence();
    v16h a3a = load_A(HA, lane);
    v16h a3b = load_A(HB, lane);

    // layer 3
    v8f oa = wmma16(a3a, W3B, z);
    v8f ob = wmma16(a3b, W3B, z);
    bias_relu(oa, b3n); bias_relu(ob, b3n);

    // masked per-sample agent sum (exclude agent == sel)
    float accs = 0.f;
#pragma unroll
    for (int v = 0; v < 8; ++v) {
      int agA = mb + v;
      int agB = 16 + mb + v;
      accs += (agA != sel) ? oa[v] : 0.f;
      accs += (agB != sel) ? ob[v] : 0.f;
    }
    float tot = accs + __shfl_xor(accs, 16, 32);   // fold lane halves
    if (lane < 16) Gsw[sl * 32 + 16 + lane] = (_Float16)tot;
  }

  // ---- gating head: 32->32 relu, 32->2 (N zero-padded to 16) ----
  v16h gW1B0 = load_Btile(gW1, 32, 32, 32, 0, lane);
  v16h gW1B1 = load_Btile(gW1, 32, 32, 32, 16, lane);
  v16h gW2B  = load_Btile(gW2, 2, 32, 2, 0, lane);
  float gb1n0 = bias_n(gb1, 32, 0, lane), gb1n1 = bias_n(gb1, 32, 16, lane);
  float gb2n  = bias_n(gb2, 2, 0, lane);

  lds_fence();
  v16h ag = load_A(Gsw, lane);
  v8f h0 = wmma16(ag, gW1B0, z);
  v8f h1 = wmma16(ag, gW1B1, z);
  bias_relu(h0, gb1n0); bias_relu(h1, gb1n1);
  lds_fence();
  store_C(HA, h0, 0, lane); store_C(HA, h1, 16, lane);
  lds_fence();
  v16h aq = load_A(HA, lane);
  v8f q = wmma16(aq, gW2B, z);
#pragma unroll
  for (int i = 0; i < 8; ++i) q[i] += gb2n;

  // ---- gather q[:, action] ----
  int n = lane & 15;
#pragma unroll
  for (int v = 0; v < 8; ++v) {
    int M = mb + v;
    int act = acts[w][M];
    if (n == act) out[s0 + M] = q[v];
  }
}

extern "C" void kernel_launch(void* const* d_in, const int* in_sizes, int n_in,
                              void* d_out, int out_size, void* d_ws,
                              size_t ws_size, hipStream_t stream) {
  const float* x   = (const float*)d_in[0];
  const int* selp  = (const int*)d_in[1];
  const float* oW1 = (const float*)d_in[2];
  const float* ob1 = (const float*)d_in[3];
  const float* oW2 = (const float*)d_in[4];
  const float* ob2 = (const float*)d_in[5];
  const float* oW3 = (const float*)d_in[6];
  const float* ob3 = (const float*)d_in[7];
  const float* sW1 = (const float*)d_in[8];
  const float* sb1 = (const float*)d_in[9];
  const float* sW2 = (const float*)d_in[10];
  const float* sb2 = (const float*)d_in[11];
  const float* sW3 = (const float*)d_in[12];
  const float* sb3 = (const float*)d_in[13];
  const float* gW1 = (const float*)d_in[14];
  const float* gb1 = (const float*)d_in[15];
  const float* gW2 = (const float*)d_in[16];
  const float* gb2 = (const float*)d_in[17];
  float* out = (float*)d_out;

  int Btot = in_sizes[0] / 128;            // A*D = 128 floats per sample
  int blocks = (Btot + 127) / 128;         // 8 waves * 16 samples per block
  dqnn_kernel<<<blocks, 256, 0, stream>>>(x, selp, oW1, ob1, oW2, ob2, oW3,
                                          ob3, sW1, sb1, sW2, sb2, sW3, sb3,
                                          gW1, gb1, gW2, gb2, out, Btot);
}
